// SDCN_45535243272745
// MI455X (gfx1250) — matrix operand order repaired and checked by reading, not compiled, not run on hardware
//
#include <hip/hip_runtime.h>
#include <hip/hip_bf16.h>

#define N_NODES 50000
#define E_EDGES 400000
#define SIGMA_F 0.3f

typedef __attribute__((ext_vector_type(16))) __bf16 v16bf;
typedef __attribute__((ext_vector_type(8)))  __bf16 v8bf;
typedef __attribute__((ext_vector_type(8)))  float  v8f;

__device__ __forceinline__ v8bf cvt8bf(const float* r) {
    v8bf o;
    #pragma unroll
    for (int i = 0; i < 8; ++i) o[i] = (__bf16)r[i];   // native f32->bf16 (RNE)
    return o;
}

// ---------------------------------------------------------------------------
// Fused GEMM:  C[M,N] = actOut( (alpha * actA(A1) + beta * A2) @ W + bias )
//   W is [K,N] row-major. A1/A2 are [M,K]. bias may be null.
// 256 threads = 8 wave32. Workgroup tile 128x128. Wave tile 64x32 (4x2 WMMA).
// Double-buffered LDS; global loads for tile k+1 staged in registers while
// WMMAs consume tile k; one barrier per k-step.
// ---------------------------------------------------------------------------
template<bool HAS_A2, bool RELU_A1>
__global__ __launch_bounds__(256)
void gemm_wmma(const float* __restrict__ A1, const float* __restrict__ A2,
               float alpha, float beta,
               const float* __restrict__ W, const float* __restrict__ bias,
               float* __restrict__ C, int M, int K, int N, int reluOut)
{
    __shared__ __align__(64) __bf16 Ash[2][128][32];   // [buf][m][k]
    __shared__ __align__(64) __bf16 Bsh[2][128][32];   // [buf][n][k]

    const int t    = threadIdx.x;
    const int lane = t & 31;
    const int wave = t >> 5;
    const int wm   = wave & 1;    // M half of workgroup tile
    const int wn   = wave >> 1;   // N quarter
    const int lr   = lane & 15;
    const int lh   = lane >> 4;

    const int m0 = blockIdx.y * 128;
    const int n0 = blockIdx.x * 128;

    // --- staging maps ---
    // A: thread covers row (t>>1), k-cols ac0..ac0+15 (contiguous in global)
    const int  arow  = t >> 1;
    const int  ac0   = (t & 1) << 4;
    const int  gm    = m0 + arow;
    const bool rowOK = gm < M;
    const long ro    = (long)(rowOK ? gm : 0) * K;
    // B: thread covers col (t&127), k-rows kc0..kc0+15 (coalesced across lanes)
    const int  bn    = t & 127;
    const int  kc0   = (t >> 7) << 4;
    const int  gn    = n0 + bn;
    const bool colOK = gn < N;

    float aReg[16], bReg[16];

    auto loadTile = [&](int kt) {
        const int kb = kt << 5;
        const bool fullK = (kb + 32 <= K);
        // ---- A (+ fused skip/relu) ----
        if (rowOK && fullK) {
            const float4* p4 = (const float4*)(A1 + ro + kb + ac0);
            float4 v0 = p4[0], v1 = p4[1], v2 = p4[2], v3 = p4[3];
            aReg[0]=v0.x; aReg[1]=v0.y; aReg[2]=v0.z; aReg[3]=v0.w;
            aReg[4]=v1.x; aReg[5]=v1.y; aReg[6]=v1.z; aReg[7]=v1.w;
            aReg[8]=v2.x; aReg[9]=v2.y; aReg[10]=v2.z; aReg[11]=v2.w;
            aReg[12]=v3.x; aReg[13]=v3.y; aReg[14]=v3.z; aReg[15]=v3.w;
            if (RELU_A1) {
                #pragma unroll
                for (int j = 0; j < 16; ++j) aReg[j] = fmaxf(aReg[j], 0.0f);
            }
            if (HAS_A2) {
                #pragma unroll
                for (int j = 0; j < 16; ++j) aReg[j] *= alpha;
                const float4* q4 = (const float4*)(A2 + ro + kb + ac0);
                float4 u0 = q4[0], u1 = q4[1], u2 = q4[2], u3 = q4[3];
                aReg[0]+=beta*u0.x; aReg[1]+=beta*u0.y; aReg[2]+=beta*u0.z; aReg[3]+=beta*u0.w;
                aReg[4]+=beta*u1.x; aReg[5]+=beta*u1.y; aReg[6]+=beta*u1.z; aReg[7]+=beta*u1.w;
                aReg[8]+=beta*u2.x; aReg[9]+=beta*u2.y; aReg[10]+=beta*u2.z; aReg[11]+=beta*u2.w;
                aReg[12]+=beta*u3.x; aReg[13]+=beta*u3.y; aReg[14]+=beta*u3.z; aReg[15]+=beta*u3.w;
            }
        } else {
            #pragma unroll
            for (int j = 0; j < 16; ++j) {
                const int  kg = kb + ac0 + j;
                const bool ok = rowOK && (kg < K);
                const int  kc = ok ? kg : 0;
                float a = A1[ro + kc];                 // unconditional, clamped
                if (RELU_A1) a = fmaxf(a, 0.0f);
                float v = HAS_A2 ? (alpha * a) : a;    // alpha==1 for plain GEMM
                if (HAS_A2) v += beta * A2[ro + kc];
                aReg[j] = ok ? v : 0.0f;
            }
        }
        // ---- B (coalesced dwords across lanes; offsets fold to immediates) ----
        if (colOK) {
            if (fullK) {
                const float* wp = W + (long)(kb + kc0) * N + gn;
                #pragma unroll
                for (int j = 0; j < 16; ++j) bReg[j] = wp[(long)j * N];
            } else {
                #pragma unroll
                for (int j = 0; j < 16; ++j) {
                    const int  kg = kb + kc0 + j;
                    const bool ok = kg < K;
                    const int  kc = ok ? kg : 0;
                    float v = W[(long)kc * N + gn];
                    bReg[j] = ok ? v : 0.0f;
                }
            }
        } else {
            #pragma unroll
            for (int j = 0; j < 16; ++j) bReg[j] = 0.0f;
        }
    };

    auto storeTile = [&](int p) {
        *(v8bf*)&Ash[p][arow][ac0]     = cvt8bf(aReg);
        *(v8bf*)&Ash[p][arow][ac0 + 8] = cvt8bf(aReg + 8);
        *(v8bf*)&Bsh[p][bn][kc0]       = cvt8bf(bReg);
        *(v8bf*)&Bsh[p][bn][kc0 + 8]   = cvt8bf(bReg + 8);
    };

    v8f acc[4][2];
    #pragma unroll
    for (int mm = 0; mm < 4; ++mm)
        #pragma unroll
        for (int nn = 0; nn < 2; ++nn)
            #pragma unroll
            for (int i = 0; i < 8; ++i)
                acc[mm][nn][i] = 0.0f;

    const int ksteps = (K + 31) >> 5;
    loadTile(0);
    int p = 0;
    for (int kt = 0; kt < ksteps; ++kt) {
        storeTile(p);
        __syncthreads();
        if (kt + 1 < ksteps) loadTile(kt + 1);   // overlap with WMMAs below

        v16bf bf[2];
        #pragma unroll
        for (int nn = 0; nn < 2; ++nn)
            bf[nn] = *(const v16bf*)&Bsh[p][wn * 32 + nn * 16 + lr][lh * 16];
        #pragma unroll
        for (int mm = 0; mm < 4; ++mm) {
            const __bf16* pa = &Ash[p][wm * 64 + mm * 16 + lr][lh * 16];
            v8bf lo = *(const v8bf*)pa;          // k = lh*16 + 0..7
            v8bf hi = *(const v8bf*)(pa + 8);    // k = lh*16 + 8..15
            v16bf a = __builtin_shufflevector(lo, hi,
                0, 1, 2, 3, 4, 5, 6, 7, 8, 9, 10, 11, 12, 13, 14, 15);
            #pragma unroll
            for (int nn = 0; nn < 2; ++nn)
                acc[mm][nn] = __builtin_amdgcn_wmma_f32_16x16x32_bf16(
                    false, a, false, bf[nn], (short)0, acc[mm][nn], false, false);
        }
        p ^= 1;
    }

    // ---- epilogue: bias + optional ReLU, guarded stores ----
    #pragma unroll
    for (int nn = 0; nn < 2; ++nn) {
        const int col = n0 + wn * 32 + nn * 16 + lr;
        if (col >= N) continue;
        const float b = bias ? bias[col] : 0.0f;
        #pragma unroll
        for (int mm = 0; mm < 4; ++mm) {
            #pragma unroll
            for (int r = 0; r < 8; ++r) {
                const int row = m0 + wm * 64 + mm * 16 + lh * 8 + r;
                if (row < M) {
                    float v = acc[mm][nn][r] + b;
                    if (reluOut) v = fmaxf(v, 0.0f);
                    C[(long)row * N + col] = v;
                }
            }
        }
    }
}

// ---------------------------------------------------------------------------
// conv0: x[N,3,500] -> pro_x[N,500]  (3 in-ch -> 1 out-ch, k=3, pad 1)
// ---------------------------------------------------------------------------
__global__ void conv0_kernel(const float* __restrict__ x,
                             const float* __restrict__ w,
                             const float* __restrict__ b,
                             float* __restrict__ pro)
{
    long idx = (long)blockIdx.x * blockDim.x + threadIdx.x;
    if (idx >= (long)N_NODES * 500) return;
    int n = (int)(idx / 500), l = (int)(idx % 500);
    const float* xr = x + (long)n * 1500;
    float s = b[0];
    #pragma unroll
    for (int v = 0; v < 3; ++v) {
        const float* xv = xr + v * 500;
        #pragma unroll
        for (int k = 0; k < 3; ++k) {
            int pidx = l + k - 1;
            if (pidx >= 0 && pidx < 500) s += w[v * 3 + k] * xv[pidx];
        }
    }
    pro[idx] = s;
}

// ---------------------------------------------------------------------------
// conv1: xb[N,500] -> x_bar[N,3,500]  (1 in-ch -> 3 out-ch, k=3, pad 1)
// ---------------------------------------------------------------------------
__global__ void conv1_kernel(const float* __restrict__ xb,
                             const float* __restrict__ w,
                             const float* __restrict__ b,
                             float* __restrict__ out)
{
    long idx = (long)blockIdx.x * blockDim.x + threadIdx.x;
    if (idx >= (long)N_NODES * 3 * 500) return;
    int n = (int)(idx / 1500);
    int rem = (int)(idx % 1500);
    int c = rem / 500, l = rem % 500;
    const float* xr = xb + (long)n * 500;
    float s = b[c];
    #pragma unroll
    for (int k = 0; k < 3; ++k) {
        int pidx = l + k - 1;
        if (pidx >= 0 && pidx < 500) s += w[c * 3 + k] * xr[pidx];
    }
    out[idx] = s;
}

__global__ void zero_kernel(float* __restrict__ p, long n)
{
    long i = (long)blockIdx.x * blockDim.x + threadIdx.x;
    if (i < n) p[i] = 0.0f;
}

// ---------------------------------------------------------------------------
// SpMM: out[dst[e], f] += w[e] * T[src[e], f]   (out pre-zeroed)
// ---------------------------------------------------------------------------
__global__ void spmm_kernel(const float* __restrict__ T,
                            const int* __restrict__ src,
                            const int* __restrict__ dst,
                            const float* __restrict__ ew,
                            float* __restrict__ out, int F, int fq)
{
    long idx = (long)blockIdx.x * blockDim.x + threadIdx.x;
    if (idx >= (long)E_EDGES * fq) return;
    int e  = (int)(idx / fq);
    int f0 = (int)(idx % fq) * 4;
    int s = src[e], d = dst[e];
    float w = ew[e];
    const float* tr = T + (long)s * F;
    float* orow = out + (long)d * F;
    #pragma unroll
    for (int j = 0; j < 4; ++j) {
        int f = f0 + j;
        if (f < F) atomicAdd(&orow[f], w * tr[f]);
    }
}

__global__ void softmax10_kernel(float* __restrict__ p)
{
    int n = blockIdx.x * blockDim.x + threadIdx.x;
    if (n >= N_NODES) return;
    float* r = p + (long)n * 10;
    float m = r[0];
    #pragma unroll
    for (int i = 1; i < 10; ++i) m = fmaxf(m, r[i]);
    float e[10], s = 0.0f;
    #pragma unroll
    for (int i = 0; i < 10; ++i) { e[i] = __expf(r[i] - m); s += e[i]; }
    float inv = 1.0f / s;
    #pragma unroll
    for (int i = 0; i < 10; ++i) r[i] = e[i] * inv;
}

// ---------------------------------------------------------------------------
extern "C" void kernel_launch(void* const* d_in, const int* in_sizes, int n_in,
                              void* d_out, int out_size, void* d_ws, size_t ws_size,
                              hipStream_t stream)
{
    const float* x       = (const float*)d_in[0];
    const int*   eidx    = (const int*)  d_in[1];
    const float* ew      = (const float*)d_in[2];
    const float* conv0_w = (const float*)d_in[3];
    const float* conv0_b = (const float*)d_in[4];
    const float* conv1_w = (const float*)d_in[5];
    const float* conv1_b = (const float*)d_in[6];
    const float* enc1_w  = (const float*)d_in[7];  const float* enc1_b = (const float*)d_in[8];
    const float* enc2_w  = (const float*)d_in[9];  const float* enc2_b = (const float*)d_in[10];
    const float* enc3_w  = (const float*)d_in[11]; const float* enc3_b = (const float*)d_in[12];
    const float* z_w     = (const float*)d_in[13]; const float* z_b    = (const float*)d_in[14];
    const float* dec1_w  = (const float*)d_in[15]; const float* dec1_b = (const float*)d_in[16];
    const float* dec2_w  = (const float*)d_in[17]; const float* dec2_b = (const float*)d_in[18];
    const float* dec3_w  = (const float*)d_in[19]; const float* dec3_b = (const float*)d_in[20];
    const float* xbar_w  = (const float*)d_in[21]; const float* xbar_b = (const float*)d_in[22];
    const float* gnn1_w  = (const float*)d_in[23];
    const float* gnn2_w  = (const float*)d_in[24];
    const float* gnn3_w  = (const float*)d_in[25];
    const float* gnn4_w  = (const float*)d_in[26];
    const float* fc_w    = (const float*)d_in[27]; const float* fc_b   = (const float*)d_in[28];

    const int* src = eidx;
    const int* dst = eidx + E_EDGES;

    // output slots: x_bar[N,3,500] | predict[N,10] | z[N,10] | pro_x[N,500]
    float* out    = (float*)d_out;
    float* xbar_o = out;
    float* pred_o = out + 75000000L;
    float* z_o    = out + 75500000L;
    float* prox_o = out + 76000000L;

    // workspace: enc_h1 | enc_h2 | enc_h3 | ping | pong | small1 | small2
    float* ws = (float*)d_ws;
    float* e1 = ws;
    float* e2 = ws + 25000000L;
    float* e3 = ws + 50000000L;
    float* bA = ws + 75000000L;
    float* bB = ws + 100000000L;
    float* s1 = ws + 125000000L;   // [N,10]
    float* s2 = ws + 125500000L;   // [N,10]

    const int NB = 256;
    auto gemm = [&](const float* A1, const float* A2, float alpha, float beta,
                    int reluA1, const float* W, const float* bias, float* C,
                    int M, int K, int N, int reluOut) {
        dim3 g((unsigned)((N + 127) / 128), (unsigned)((M + 127) / 128));
        if (A2) {
            if (reluA1)
                gemm_wmma<true, true><<<g, dim3(256), 0, stream>>>(
                    A1, A2, alpha, beta, W, bias, C, M, K, N, reluOut);
            else
                gemm_wmma<true, false><<<g, dim3(256), 0, stream>>>(
                    A1, A2, alpha, beta, W, bias, C, M, K, N, reluOut);
        } else {
            gemm_wmma<false, false><<<g, dim3(256), 0, stream>>>(
                A1, A2, alpha, beta, W, bias, C, M, K, N, reluOut);
        }
    };
    auto zero = [&](float* p, long n) {
        zero_kernel<<<(unsigned)((n + NB - 1) / NB), NB, 0, stream>>>(p, n);
    };
    auto spmm = [&](const float* T, float* O, int F) {
        int fq = (F + 3) / 4;
        long total = (long)E_EDGES * fq;
        spmm_kernel<<<(unsigned)((total + NB - 1) / NB), NB, 0, stream>>>(
            T, src, dst, ew, O, F, fq);
    };

    // ---- AE encoder ----
    long c0n = (long)N_NODES * 500;
    conv0_kernel<<<(unsigned)((c0n + NB - 1) / NB), NB, 0, stream>>>(x, conv0_w, conv0_b, prox_o);
    gemm(prox_o, nullptr, 1.0f, 0.0f, 0, enc1_w, enc1_b, e1, N_NODES, 500, 500, 1);
    gemm(e1,     nullptr, 1.0f, 0.0f, 0, enc2_w, enc2_b, e2, N_NODES, 500, 500, 1);
    gemm(e2,     nullptr, 1.0f, 0.0f, 0, enc3_w, enc3_b, e3, N_NODES, 500, 500, 1);
    gemm(e3,     nullptr, 1.0f, 0.0f, 0, z_w,    z_b,    z_o, N_NODES, 500, 10, 0);

    // ---- AE decoder ----
    gemm(z_o, nullptr, 1.0f, 0.0f, 0, dec1_w, dec1_b, bA, N_NODES, 10,  500, 1);
    gemm(bA,  nullptr, 1.0f, 0.0f, 0, dec2_w, dec2_b, bB, N_NODES, 500, 500, 1);
    gemm(bB,  nullptr, 1.0f, 0.0f, 0, dec3_w, dec3_b, bA, N_NODES, 500, 500, 1);
    gemm(bA,  nullptr, 1.0f, 0.0f, 0, xbar_w, xbar_b, bB, N_NODES, 500, 500, 1);
    long c1n = (long)N_NODES * 3 * 500;
    conv1_kernel<<<(unsigned)((c1n + NB - 1) / NB), NB, 0, stream>>>(bB, conv1_w, conv1_b, xbar_o);

    // ---- GCN with weighted skip fusion (relu of spmm fused into next A-load) ----
    const float Af = 1.0f - SIGMA_F, Sf = SIGMA_F;
    gemm(prox_o, nullptr, 1.0f, 0.0f, 0, gnn1_w, nullptr, bA, N_NODES, 500, 500, 0);
    zero(bB, (long)N_NODES * 500);
    spmm(bA, bB, 500);                                   // bB = h1_raw
    gemm(bB, e1, Af, Sf, 1, gnn2_w, nullptr, bA, N_NODES, 500, 500, 0);
    zero(bB, (long)N_NODES * 500);
    spmm(bA, bB, 500);                                   // bB = h2_raw
    gemm(bB, e2, Af, Sf, 1, gnn3_w, nullptr, bA, N_NODES, 500, 500, 0);
    zero(bB, (long)N_NODES * 500);
    spmm(bA, bB, 500);                                   // bB = h3_raw
    gemm(bB, e3, Af, Sf, 1, gnn4_w, nullptr, s1, N_NODES, 500, 10, 0);
    zero(s2, (long)N_NODES * 10);
    spmm(s1, s2, 10);                                    // s2 = h4 (no relu)
    gemm(s2, z_o, Af, Sf, 0, fc_w, fc_b, pred_o, N_NODES, 10, 10, 0);
    softmax10_kernel<<<(N_NODES + NB - 1) / NB, NB, 0, stream>>>(pred_o);
}
